// Network_32865089749313
// MI455X (gfx1250) — matrix operand order repaired and checked by reading, not compiled, not run on hardware
//
#include <hip/hip_runtime.h>
#include <hip/hip_bf16.h>

// ---------------------------------------------------------------------------
// MVSNet-style pipeline for MI455X (gfx1250, wave32).
// Heavy stage (3D conv 32->8, K=864, 9.1 GFLOP of 11.5 total) is mapped onto
// v_wmma_f32_16x16x32_f16. The f16 cost volume is stored with a 1-voxel
// zero halo so the WMMA K-loop has NO bounds checks: 27 taps become pure
// immediate-offset global_load_b128 + ds_load_b128 + v_wmma streams.
// ---------------------------------------------------------------------------

typedef _Float16 h16v __attribute__((ext_vector_type(16)));
typedef _Float16 h8v  __attribute__((ext_vector_type(8)));
typedef float    f8v  __attribute__((ext_vector_type(8)));

namespace {
constexpr int NV = 5;
constexpr int HIMG = 512, WIMG = 640;
constexpr int CFN = 32;          // feature channels
constexpr int ND  = 128;         // depth hypotheses
constexpr int HF = 64, WF = 80;  // feature resolution
// padded (halo) cost volume dims
constexpr int DP = ND + 2, HP = HF + 2, WP = WF + 2;
constexpr long COSTH_HALVES = (long)DP * HP * WP * CFN;
}

// ---------------------------------------------------------------------------
// Setup: per-view homography params  uvw = (Kf*Rrel*Kf^-1) pix * depth + Kf*trel
// and reg_w1 transposed to f16 B-matrix table wT[tap][n][k] (n=out ch, k=in ch)
// ---------------------------------------------------------------------------
__device__ inline void mat3mul(float* o, const float* a, const float* b) {
  for (int r = 0; r < 3; ++r)
    for (int c = 0; c < 3; ++c)
      o[r*3+c] = a[r*3+0]*b[0*3+c] + a[r*3+1]*b[1*3+c] + a[r*3+2]*b[2*3+c];
}

__global__ void __launch_bounds__(256) setup_kernel(
    const float* __restrict__ K, const float* __restrict__ poses,
    const float* __restrict__ w1, float* __restrict__ trans,
    _Float16* __restrict__ wT)
{
  int tid = threadIdx.x;
  // wT[tap][n][k]: B[k][n] per tap; n>=8 padded with zero
  for (int i = tid; i < 27*16*32; i += 256) {
    int tap = i >> 9; int rem = i & 511; int n = rem >> 5; int k = rem & 31;
    int kz = tap / 9, ky = (tap / 3) % 3, kx = tap % 3;
    float val = 0.f;
    if (n < 8) val = w1[(((n*32 + k)*3 + kz)*3 + ky)*3 + kx];
    wT[i] = (_Float16)val;
  }
  if (tid == 0) {
    float fx = K[0]*0.125f, cx = K[2]*0.125f, fy = K[4]*0.125f, cy = K[5]*0.125f;
    float Kf[9] = {fx,0,cx, 0,fy,cy, 0,0,1};
    float iK[9] = {1.f/fx,0,-cx/fx, 0,1.f/fy,-cy/fy, 0,0,1};
    float R0[9], t0[3];
    for (int r = 0; r < 3; ++r) {
      for (int c = 0; c < 3; ++c) R0[r*3+c] = poses[r*4+c];
      t0[r] = poses[r*4+3];
    }
    for (int v = 1; v < NV; ++v) {
      const float* P = poses + v*16;
      float Rv[9], tv[3];
      for (int r = 0; r < 3; ++r) {
        for (int c = 0; c < 3; ++c) Rv[r*3+c] = P[r*4+c];
        tv[r] = P[r*4+3];
      }
      float Rr[9];  // Rv * R0^T
      for (int r = 0; r < 3; ++r)
        for (int c = 0; c < 3; ++c)
          Rr[r*3+c] = Rv[r*3+0]*R0[c*3+0] + Rv[r*3+1]*R0[c*3+1] + Rv[r*3+2]*R0[c*3+2];
      float tr[3];
      for (int r = 0; r < 3; ++r)
        tr[r] = tv[r] - (Rr[r*3+0]*t0[0] + Rr[r*3+1]*t0[1] + Rr[r*3+2]*t0[2]);
      float T1[9], A[9];
      mat3mul(T1, Kf, Rr); mat3mul(A, T1, iK);
      float* o = trans + (v-1)*12;
      for (int j = 0; j < 9; ++j) o[j] = A[j];
      for (int r = 0; r < 3; ++r)
        o[9+r] = Kf[r*3+0]*tr[0] + Kf[r*3+1]*tr[1] + Kf[r*3+2]*tr[2];
    }
  }
}

// ---------------------------------------------------------------------------
// Vectorized zero fill (graph-capture-safe halo init for padded cost volume)
// ---------------------------------------------------------------------------
__global__ void __launch_bounds__(256) fill_zero_kernel(uint4* __restrict__ p, long n16)
{
  long i = (long)blockIdx.x*256 + threadIdx.x;
  if (i < n16) p[i] = make_uint4(0u, 0u, 0u, 0u);
}

// ---------------------------------------------------------------------------
// Encoder: stride-2 SAME conv (XLA: pad_lo=0, pad_hi=1  ->  in = 2*out + k)
// ---------------------------------------------------------------------------
template<int CI, int CO, int HIN, int WIN, bool RELU, bool CHLAST>
__global__ void __launch_bounds__(256) conv2d_s2(
    const float* __restrict__ in, const float* __restrict__ w,
    const float* __restrict__ b, float* __restrict__ out)
{
  constexpr int HO = HIN/2, WO = WIN/2;
  long idx = (long)blockIdx.x*256 + threadIdx.x;
  if (idx >= (long)NV*CO*HO*WO) return;
  int o, x, y, v;
  if (CHLAST) { o = idx % CO; x = (idx/CO) % WO; y = (idx/((long)CO*WO)) % HO; v = idx/((long)CO*WO*HO); }
  else        { x = idx % WO; y = (idx/WO) % HO; o = (idx/((long)WO*HO)) % CO; v = idx/((long)WO*HO*CO); }
  const float* ip = in + (long)v*CI*HIN*WIN;
  float acc = b[o];
  for (int ci = 0; ci < CI; ++ci)
    for (int ky = 0; ky < 3; ++ky) {
      int iy = 2*y + ky; if (iy >= HIN) continue;
      for (int kx = 0; kx < 3; ++kx) {
        int ix = 2*x + kx; if (ix >= WIN) continue;
        acc += ip[((long)ci*HIN + iy)*WIN + ix] * w[((o*CI + ci)*3 + ky)*3 + kx];
      }
    }
  if (RELU) acc = fmaxf(acc, 0.f);
  if (CHLAST) out[(((long)v*HO + y)*WO + x)*CO + o] = acc;
  else        out[(((long)v*CO + o)*HO + y)*WO + x] = acc;
}

// ---------------------------------------------------------------------------
// Plane-sweep variance cost volume. feats channel-last (v,y,x,c).
// Writes f16 channel-last into the INTERIOR of the padded volume
// costh[(d+1, y+1, x+1, c)]; halo stays zero from fill_zero_kernel.
// ---------------------------------------------------------------------------
__global__ void __launch_bounds__(256) costvol_kernel(
    const float* __restrict__ feats, const float* __restrict__ trans,
    _Float16* __restrict__ costh)
{
  int idx = blockIdx.x*256 + threadIdx.x;               // (d*HF + y)*WF + x
  int x = idx % WF; int y = (idx / WF) % HF; int d = idx / (WF*HF);
  float inv_d = 2.0f + (0.2f - 2.0f) * (float)d / 127.0f;
  float depth = 1.0f / inv_d;
  float s[CFN], ss[CFN];
  const float* f0 = feats + ((long)y*WF + x)*CFN;       // reference view v=0
  #pragma unroll
  for (int c = 0; c < CFN; ++c) { float f = f0[c]; s[c] = f; ss[c] = f*f; }
  float px = (float)x, py = (float)y;
  for (int v = 1; v < NV; ++v) {
    const float* T = trans + (v-1)*12;
    float qx = T[0]*px + T[1]*py + T[2];
    float qy = T[3]*px + T[4]*py + T[5];
    float qz = T[6]*px + T[7]*py + T[8];
    float u  = qx*depth + T[9];
    float vv = qy*depth + T[10];
    float z  = qz*depth + T[11];
    float valid = (z > 0.001f) ? 1.f : 0.f;
    float zc = fmaxf(z, 0.001f);
    float fu = u / zc, fv = vv / zc;
    float x0f = floorf(fu), y0f = floorf(fv);
    float wx = fu - x0f, wy = fv - y0f;
    int x0 = (int)x0f, y0 = (int)y0f, x1 = x0 + 1, y1 = y0 + 1;
    float m00 = (x0 >= 0 && x0 < WF && y0 >= 0 && y0 < HF) ? 1.f : 0.f;
    float m01 = (x1 >= 0 && x1 < WF && y0 >= 0 && y0 < HF) ? 1.f : 0.f;
    float m10 = (x0 >= 0 && x0 < WF && y1 >= 0 && y1 < HF) ? 1.f : 0.f;
    float m11 = (x1 >= 0 && x1 < WF && y1 >= 0 && y1 < HF) ? 1.f : 0.f;
    int xc0 = min(max(x0,0),WF-1), xc1 = min(max(x1,0),WF-1);
    int yc0 = min(max(y0,0),HF-1), yc1 = min(max(y1,0),HF-1);
    float w00 = (1.f-wx)*(1.f-wy)*m00*valid;
    float w01 =      wx *(1.f-wy)*m01*valid;
    float w10 = (1.f-wx)*     wy *m10*valid;
    float w11 =      wx *     wy *m11*valid;
    const float* base = feats + (long)v*HF*WF*CFN;
    const float* p00 = base + ((long)yc0*WF + xc0)*CFN;
    const float* p01 = base + ((long)yc0*WF + xc1)*CFN;
    const float* p10 = base + ((long)yc1*WF + xc0)*CFN;
    const float* p11 = base + ((long)yc1*WF + xc1)*CFN;
    #pragma unroll
    for (int c = 0; c < CFN; ++c) {
      float val = w00*p00[c] + w01*p01[c] + w10*p10[c] + w11*p11[c];
      s[c] += val; ss[c] += val*val;
    }
  }
  _Float16* out = costh + ((((long)(d+1))*HP + (y+1))*WP + (x+1))*CFN;
  #pragma unroll
  for (int c = 0; c < CFN; ++c) {
    float mean = s[c]*0.2f;
    out[c] = (_Float16)(ss[c]*0.2f - mean*mean);
  }
}

// ---------------------------------------------------------------------------
// 3D conv 32->8 (3x3x3) as implicit GEMM on WMMA f16 (16x16x32).
// Per wave: M=16 x-positions, N=16 (8 real out channels), K = 27 taps x 32ch.
// Halo-padded input -> no bounds checks; all 27 taps are immediate-offset
// loads from one hoisted base pointer.
// A layout per ISA (16-bit A 16x32): lanes 0-15 K0-7/K16-23; lanes 16-31
// K8-15/K24-31.  B from LDS wT[tap][n][k]: lanes 0-15 K0-7/K16-23, 16-31 K8-15/K24-31.
// C/D: c[j] -> M = j + 8*(lane>=16), N = lane&15.
// ---------------------------------------------------------------------------
__global__ void __launch_bounds__(128) reg1_wmma_kernel(
    const _Float16* __restrict__ costh, const _Float16* __restrict__ wT,
    const float* __restrict__ b1, _Float16* __restrict__ r1)
{
  __shared__ _Float16 lw[27*16*32];
  const int tid = threadIdx.x;
  for (int i = tid; i < 27*16*32; i += 128) lw[i] = wT[i];
  __syncthreads();

  int lane = tid & 31;
  long tileId = (long)blockIdx.x*4 + (tid >> 5);   // 40960 tiles = 5*64*128
  int xt = tileId % 5; long rest = tileId / 5;
  int y = rest % HF; int d = rest / HF;
  int x0 = xt * 16;

  int n = lane & 15;             // A row M / B column N / out channel
  int hiHalf = lane >> 4;
  int aBase = hiHalf ? 8 : 0;    // A: K = aBase..aBase+7 and aBase+16..aBase+23
  int kbB   = hiHalf ? 8 : 0;    // B: K = kbB..kbB+7 and kbB+16..kbB+23
  int x = x0 + n;

  // base pointer at padded tap (dz,dy,dx)=(0,0,0): unpadded (d-1,y-1,x-1)
  const _Float16* pbase = costh + ((((long)d)*HP + y)*WP + x)*CFN + aBase;
  const _Float16* wbase = &lw[(lane & 15)*32 + kbB];

  f8v acc = {};
  #pragma unroll
  for (int dz = 0; dz < 3; ++dz) {
    #pragma unroll
    for (int dy = 0; dy < 3; ++dy) {
      #pragma unroll
      for (int dx = 0; dx < 3; ++dx) {
        int tap = (dz*3 + dy)*3 + dx;
        const _Float16* p = pbase + (((long)dz*HP + dy)*WP + dx)*CFN;
        h8v lo = *(const h8v*)(p);
        h8v hi = *(const h8v*)(p + 16);
        h16v a;
        #pragma unroll
        for (int i = 0; i < 8; ++i) { a[i] = lo[i]; a[i+8] = hi[i]; }
        const _Float16* wp = wbase + tap*16*32;
        h8v blo = *(const h8v*)(wp);
        h8v bhi = *(const h8v*)(wp + 16);
        h16v b;
        #pragma unroll
        for (int i = 0; i < 8; ++i) { b[i] = blo[i]; b[i+8] = bhi[i]; }
        acc = __builtin_amdgcn_wmma_f32_16x16x32_f16(
            false, a, false, b, (short)0, acc, false, false);
      }
    }
  }
  if (n < 8) {
    float bias = b1[n];
    #pragma unroll
    for (int j = 0; j < 8; ++j) {
      int M = j + 8*hiHalf;
      float vv = fmaxf(acc[j] + bias, 0.f);
      r1[((((long)d*HF + y)*WF + (x0 + M)))*8 + n] = (_Float16)vv;
    }
  }
}

// ---------------------------------------------------------------------------
// 3D conv 8->1 (3x3x3), direct. r1 channel-last f16 (d,y,x,8).
// ---------------------------------------------------------------------------
__global__ void __launch_bounds__(256) reg2_kernel(
    const _Float16* __restrict__ r1, const float* __restrict__ w2,
    const float* __restrict__ b2, float* __restrict__ cost2)
{
  int idx = blockIdx.x*256 + threadIdx.x;
  int x = idx % WF; int y = (idx / WF) % HF; int d = idx / (WF*HF);
  float acc = b2[0];
  for (int kz = 0; kz < 3; ++kz) {
    int dd = d + kz - 1; if ((unsigned)dd >= (unsigned)ND) continue;
    for (int ky = 0; ky < 3; ++ky) {
      int yy = y + ky - 1; if ((unsigned)yy >= (unsigned)HF) continue;
      for (int kx = 0; kx < 3; ++kx) {
        int xx = x + kx - 1; if ((unsigned)xx >= (unsigned)WF) continue;
        const _Float16* rp = r1 + (((long)dd*HF + yy)*WF + xx)*8;
        #pragma unroll
        for (int c = 0; c < 8; ++c)
          acc += (float)rp[c] * w2[((c*3 + kz)*3 + ky)*3 + kx];
      }
    }
  }
  cost2[idx] = acc;
}

// ---------------------------------------------------------------------------
// Softmax over depth + regressed depth + sliding-window-4 max confidence.
// sum4[i] = p[i-1]+p[i]+p[i+1]+p[i+2]  ->  ring buffer over d = i+2 in [2,129].
// ---------------------------------------------------------------------------
__global__ void __launch_bounds__(256) depth_stats_kernel(
    const float* __restrict__ cost2, float* __restrict__ reglow,
    float* __restrict__ conflow)
{
  int idx = blockIdx.x*256 + threadIdx.x;   // y*WF + x, 5120 total
  float mx = -1e30f;
  for (int d = 0; d < ND; ++d) mx = fmaxf(mx, cost2[(long)d*HF*WF + idx]);
  float sum = 0.f;
  for (int d = 0; d < ND; ++d) sum += expf(cost2[(long)d*HF*WF + idx] - mx);
  float inv = 1.f / sum;
  float reg = 0.f, conf = 0.f;
  float p1 = 0.f, p2 = 0.f, p3 = 0.f;
  for (int d = 0; d < ND + 2; ++d) {
    float p = 0.f;
    if (d < ND) {
      p = expf(cost2[(long)d*HF*WF + idx] - mx) * inv;
      float invd = 2.0f + (0.2f - 2.0f) * (float)d / 127.0f;
      reg += p / invd;
    }
    float s4 = p + p1 + p2 + p3;
    if (d >= 2) conf = fmaxf(conf, s4);
    p3 = p2; p2 = p1; p1 = p;
  }
  reglow[idx] = reg; conflow[idx] = conf;
}

// ---------------------------------------------------------------------------
// 8x bilinear upsample (half-pixel centers, edge clamp): conf -> d_out[0..],
// regressed -> upreg workspace.
// ---------------------------------------------------------------------------
__global__ void __launch_bounds__(256) upsample_kernel(
    const float* __restrict__ reglow, const float* __restrict__ conflow,
    float* __restrict__ upreg, float* __restrict__ outConf)
{
  int idx = blockIdx.x*256 + threadIdx.x;
  int ox = idx % WIMG; int oy = idx / WIMG;
  float sx = ((float)ox + 0.5f)*0.125f - 0.5f;
  float sy = ((float)oy + 0.5f)*0.125f - 0.5f;
  float x0f = floorf(sx), y0f = floorf(sy);
  float wx = sx - x0f, wy = sy - y0f;
  int x0 = min(max((int)x0f, 0), WF-1), x1 = min(max((int)x0f + 1, 0), WF-1);
  int y0 = min(max((int)y0f, 0), HF-1), y1 = min(max((int)y0f + 1, 0), HF-1);
  float w00 = (1.f-wx)*(1.f-wy), w01 = wx*(1.f-wy);
  float w10 = (1.f-wx)*wy,       w11 = wx*wy;
  int i00 = y0*WF + x0, i01 = y0*WF + x1, i10 = y1*WF + x0, i11 = y1*WF + x1;
  upreg[idx]   = w00*reglow[i00] + w01*reglow[i01] + w10*reglow[i10] + w11*reglow[i11];
  outConf[idx] = w00*conflow[i00] + w01*conflow[i01] + w10*conflow[i10] + w11*conflow[i11];
}

// ---------------------------------------------------------------------------
// Refinement conv1: [img0(3) ; upsampled depth] -> 32, 3x3, relu.
// hidden stored channel-last f16 so conv2 reads contiguous 32-half vectors.
// ---------------------------------------------------------------------------
__global__ void __launch_bounds__(256) refine1_kernel(
    const float* __restrict__ img0, const float* __restrict__ upreg,
    const float* __restrict__ w, const float* __restrict__ b,
    _Float16* __restrict__ hidden)
{
  long idx = (long)blockIdx.x*256 + threadIdx.x;
  int o = idx % 32; int x = (idx/32) % WIMG; int y = idx / ((long)32*WIMG);
  float acc = b[o];
  for (int c = 0; c < 4; ++c)
    for (int ky = 0; ky < 3; ++ky) {
      int iy = y + ky - 1; if ((unsigned)iy >= (unsigned)HIMG) continue;
      for (int kx = 0; kx < 3; ++kx) {
        int ix = x + kx - 1; if ((unsigned)ix >= (unsigned)WIMG) continue;
        float val = (c < 3) ? img0[((long)c*HIMG + iy)*WIMG + ix]
                            : upreg[(long)iy*WIMG + ix];
        acc += val * w[((o*4 + c)*3 + ky)*3 + kx];
      }
    }
  hidden[((long)y*WIMG + x)*32 + o] = (_Float16)fmaxf(acc, 0.f);
}

__global__ void __launch_bounds__(256) refine2_kernel(
    const _Float16* __restrict__ hidden, const float* __restrict__ upreg,
    const float* __restrict__ w, const float* __restrict__ b,
    float* __restrict__ outRefined)
{
  int idx = blockIdx.x*256 + threadIdx.x;
  int x = idx % WIMG; int y = idx / WIMG;
  float acc = b[0];
  for (int ky = 0; ky < 3; ++ky) {
    int iy = y + ky - 1; if ((unsigned)iy >= (unsigned)HIMG) continue;
    for (int kx = 0; kx < 3; ++kx) {
      int ix = x + kx - 1; if ((unsigned)ix >= (unsigned)WIMG) continue;
      const _Float16* hp = hidden + ((long)iy*WIMG + ix)*32;
      #pragma unroll
      for (int c = 0; c < 32; ++c)
        acc += (float)hp[c] * w[(c*3 + ky)*3 + kx];
    }
  }
  outRefined[idx] = upreg[idx] + acc;
}

// ---------------------------------------------------------------------------
extern "C" void kernel_launch(void* const* d_in, const int* in_sizes, int n_in,
                              void* d_out, int out_size, void* d_ws, size_t ws_size,
                              hipStream_t stream)
{
  (void)in_sizes; (void)n_in; (void)out_size; (void)ws_size;
  const float* images = (const float*)d_in[0];
  const float* K      = (const float*)d_in[1];
  const float* poses  = (const float*)d_in[2];
  const float* ew1 = (const float*)d_in[3];  const float* eb1 = (const float*)d_in[4];
  const float* ew2 = (const float*)d_in[5];  const float* eb2 = (const float*)d_in[6];
  const float* ew3 = (const float*)d_in[7];  const float* eb3 = (const float*)d_in[8];
  const float* rw1 = (const float*)d_in[9];  const float* rb1 = (const float*)d_in[10];
  const float* rw2 = (const float*)d_in[11]; const float* rb2 = (const float*)d_in[12];
  const float* fw1 = (const float*)d_in[13]; const float* fb1 = (const float*)d_in[14];
  const float* fw2 = (const float*)d_in[15]; const float* fb2 = (const float*)d_in[16];
  float* outp = (float*)d_out;   // [confidence (512*640) ; refined (512*640)]

  char* ws = (char*)d_ws;
  size_t off = 0;
  auto take = [&](size_t bytes) -> char* {
    char* p = ws + off; off += (bytes + 255) & ~(size_t)255; return p;
  };
  float*     trans  = (float*)    take(4*12*sizeof(float));
  _Float16*  wT     = (_Float16*) take((size_t)27*16*32*2);
  float*     f1     = (float*)    take((size_t)NV*8*256*320*4);
  float*     f2     = (float*)    take((size_t)NV*16*128*160*4);
  float*     feats  = (float*)    take((size_t)NV*HF*WF*CFN*4);
  _Float16*  costh  = (_Float16*) take((size_t)COSTH_HALVES*2);   // padded, halo=0
  _Float16*  r1     = (_Float16*) take((size_t)ND*HF*WF*8*2);
  float*     cost2  = (float*)    take((size_t)ND*HF*WF*4);
  float*     reglow = (float*)    take((size_t)HF*WF*4);
  float*     conflow= (float*)    take((size_t)HF*WF*4);
  float*     upreg  = (float*)    take((size_t)HIMG*WIMG*4);
  _Float16*  hidden = (_Float16*) take((size_t)HIMG*WIMG*32*2);

  setup_kernel<<<1, 256, 0, stream>>>(K, poses, rw1, trans, wT);

  // zero padded cost volume (halo must be zero for branch-free WMMA loop)
  long n16 = (COSTH_HALVES*2) / 16;
  fill_zero_kernel<<<(int)((n16 + 255) / 256), 256, 0, stream>>>((uint4*)costh, n16);

  conv2d_s2<3, 8, 512, 640, true,  false><<<12800, 256, 0, stream>>>(images, ew1, eb1, f1);
  conv2d_s2<8, 16, 256, 320, true,  false><<<6400,  256, 0, stream>>>(f1, ew2, eb2, f2);
  conv2d_s2<16, 32, 128, 160, false, true ><<<3200,  256, 0, stream>>>(f2, ew3, eb3, feats);

  costvol_kernel<<<2560, 256, 0, stream>>>(feats, trans, costh);

  reg1_wmma_kernel<<<10240, 128, 0, stream>>>(costh, wT, rb1, r1);
  reg2_kernel<<<2560, 256, 0, stream>>>(r1, rw2, rb2, cost2);

  depth_stats_kernel<<<20, 256, 0, stream>>>(cost2, reglow, conflow);
  upsample_kernel<<<1280, 256, 0, stream>>>(reglow, conflow, upreg, outp);

  refine1_kernel<<<40960, 256, 0, stream>>>(images, upreg, fw1, fb1, hidden);
  refine2_kernel<<<1280, 256, 0, stream>>>(hidden, upreg, fw2, fb2, outp + (size_t)HIMG*WIMG);
}
